// AttentionBlock_21784074125885
// MI455X (gfx1250) — matrix operand order repaired and checked by reading, not compiled, not run on hardware
//
#include <hip/hip_runtime.h>
#include <hip/hip_bf16.h>
#include <stdint.h>
#include <math.h>

// ---------------------------------------------------------------------------
// AttentionBlock for MI455X (gfx1250).
//   * all GEMMs on v_wmma_f32_16x16x32_bf16 (fp32 -> bf16 RNE on the fly)
//   * 128x256 block tiles, 64x64 wave tiles (16 WMMA / K-step / wave)
//   * double-buffered LDS + register prefetch (1 barrier per K-step)
//   * TDM (tensor_load_to_lds + s_wait_tensorcnt) streams the bf16 A tiles
//     (P in P@K, H in FFN2) asynchronously into LDS, overlapped with WMMA
// Workspace layout (needs >= 192 MB):
//   [0,   64MB) : S bf16 scores   (later reused as r2 fp32)
//   [64, 128MB) : P bf16 probs    (later reused as H bf16)
//   [128,192MB) : r1/x fp32       (LN1 in place)
// ---------------------------------------------------------------------------

typedef __attribute__((ext_vector_type(16))) __bf16 v16bf;
typedef __attribute__((ext_vector_type(8)))  float  v8f;

#define EMB 1024
#define LQ  2048
#define LK  2048
#define NB  8

#define BM 128
#define BN 256
#define BK 32

#if defined(__HIP_DEVICE_COMPILE__) && \
    __has_builtin(__builtin_amdgcn_tensor_load_to_lds) && \
    __has_builtin(__builtin_amdgcn_s_wait_tensorcnt)
#define USE_TDM 1
#else
#define USE_TDM 0
#endif

enum { MODE_S = 0, MODE_PV = 1, MODE_F1 = 2, MODE_F2 = 3 };

__device__ __forceinline__ uint16_t f32_to_bf16(float f) {
    uint32_t u = __builtin_bit_cast(uint32_t, f);
    u += 0x7FFFu + ((u >> 16) & 1u);          // round-to-nearest-even
    return (uint16_t)(u >> 16);
}
__device__ __forceinline__ uint32_t pk_bf16(float a, float b) {
    return (uint32_t)f32_to_bf16(a) | ((uint32_t)f32_to_bf16(b) << 16);
}

union FragAB { uint4 u[2]; v16bf v; };

#if USE_TDM
typedef __attribute__((ext_vector_type(4))) unsigned int u32x4;
typedef __attribute__((ext_vector_type(4))) int          i32x4;
typedef __attribute__((ext_vector_type(8))) int          i32x8;

// Build a 2D D# (group0 + group1) and issue TENSOR_LOAD_TO_LDS for a
// BM x BK bf16 tile from a row-major [rows][row_stride] bf16 tensor.
__device__ __forceinline__ void tdm_load_tile_bf16(const uint16_t* gsrc,
                                                   uint32_t lds_byte_off,
                                                   uint32_t rows,
                                                   uint32_t row_stride)
{
    uint64_t ga = (uint64_t)(uintptr_t)gsrc;
    u32x4 g0;
    g0[0] = 1u;                                   // count=1, user descriptor
    g0[1] = lds_byte_off;                          // LDS destination (bytes)
    g0[2] = (uint32_t)ga;                          // global_addr[31:0]
    g0[3] = (uint32_t)(ga >> 32) | (2u << 30);     // global_addr[56:32], type=2
    uint32_t td0 = row_stride;                     // tensor_dim0 (elements)
    uint32_t td1 = rows;                           // tensor_dim1 (rows)
    uint64_t st0 = (uint64_t)row_stride;           // tensor_dim0_stride
    i32x8 g1;
    g1[0] = (int)(1u << 16);                                    // data_size=2B
    g1[1] = (int)((td0 & 0xFFFFu) << 16);                       // dim0[15:0]
    g1[2] = (int)((td0 >> 16) | ((td1 & 0xFFFFu) << 16));       // dim0[31:16]|dim1[15:0]
    g1[3] = (int)((td1 >> 16) | ((uint32_t)BK << 16));          // dim1[31:16]|tile_dim0
    g1[4] = (int)(uint32_t)BM;                                  // tile_dim1 (tile_dim2=0)
    g1[5] = (int)(uint32_t)(st0 & 0xFFFFFFFFu);                 // stride0[31:0]
    g1[6] = (int)(uint32_t)(st0 >> 32);                          // stride0[47:32]
    g1[7] = 0;
    i32x4 z4 = {0, 0, 0, 0};
#if __clang_major__ >= 23
    i32x8 z8 = {0, 0, 0, 0, 0, 0, 0, 0};
    __builtin_amdgcn_tensor_load_to_lds(g0, g1, z4, z4, z8, 0);
#else
    __builtin_amdgcn_tensor_load_to_lds(g0, g1, z4, z4, 0);
#endif
}
#endif // USE_TDM

// ---------------------------------------------------------------------------
// Tiled GEMM: C[M,N] = A[M,K] * B(n,k) + mode epilogue. B consumed as [n][k]
// bf16 tiles in LDS (K rows for QK^T, W rows for x@W^T, transpose-on-store
// for P@K where memory is K[k][n]).
// ---------------------------------------------------------------------------
template<int MODE>
__global__ __launch_bounds__(256)
void gemm_wmma_kernel(const void* __restrict__ Abase,
                      const void* __restrict__ Bbase,
                      void*       __restrict__ Obase,
                      const float* __restrict__ extra0,   // key_mask / residual / bias
                      const float* __restrict__ extra1,   // second residual (F2)
                      int M, int N, int K, float scale)
{
    constexpr bool A_IS_BF16   = (MODE == MODE_PV || MODE == MODE_F2);
    constexpr bool B_TRANSPOSE = (MODE == MODE_PV);
    constexpr bool A_TDM       = (USE_TDM != 0) && A_IS_BF16;

    __shared__ __align__(16) uint16_t sA[2][BM * BK];
    __shared__ __align__(16) uint16_t sB[2][BN * BK];

    const int tid  = threadIdx.x;
    const int wid  = tid >> 5;
    const int lane = tid & 31;
    const int wm   = wid >> 2;          // 0..1  -> 64 rows each
    const int wn   = wid & 3;           // 0..3  -> 64 cols each
    const int lhalf = lane & 15;
    const int lhi   = lane >> 4;

    const int bz = blockIdx.z;
    const int m0 = blockIdx.y * BM;
    const int n0 = blockIdx.x * BN;

    const size_t zA = (size_t)bz * (size_t)M * (size_t)K;
    const size_t zB = (size_t)bz * (size_t)N * (size_t)K;
    const size_t zO = (size_t)bz * (size_t)M * (size_t)N;

    const float*    Af = (const float*)Abase    + zA;
    const uint16_t* Ab = (const uint16_t*)Abase + zA;
    const float*    Bf = (const float*)Bbase    + zB;

    v8f zero = {};
    v8f acc[4][4];
#pragma unroll
    for (int i = 0; i < 4; ++i)
#pragma unroll
        for (int j = 0; j < 4; ++j) acc[i][j] = zero;

    // ---- prefetch register staging --------------------------------------
    const int a_row0 = tid >> 3;        // fp32 A: 8 lanes per 32-float row
    const int a_col  = (tid & 7) * 4;
    float4 prA[4];
    const int ab_row0 = tid >> 2;       // bf16 A fallback: 4 lanes per row
    const int ab_col  = (tid & 3) * 8;
    uint4 prAb[2];
    const int b_row0 = tid >> 3;        // B direct
    const int b_col  = (tid & 7) * 4;
    const int bt_k   = tid >> 3;        // B transpose: k-row
    const int bt_n0  = (tid & 7) * 32;
    float4 prB[8];

    auto loadA = [&](int k0) {
        if constexpr (A_IS_BF16) {
#pragma unroll
            for (int it = 0; it < 2; ++it)
                prAb[it] = *(const uint4*)(Ab + (size_t)(m0 + ab_row0 + it * 64) * K + (k0 + ab_col));
        } else {
#pragma unroll
            for (int it = 0; it < 4; ++it)
                prA[it] = *(const float4*)(Af + (size_t)(m0 + a_row0 + it * 32) * K + (k0 + a_col));
        }
    };
    auto storeA = [&](uint16_t* As) {
        if constexpr (A_IS_BF16) {
#pragma unroll
            for (int it = 0; it < 2; ++it)
                *(uint4*)&As[(ab_row0 + it * 64) * BK + ab_col] = prAb[it];
        } else {
#pragma unroll
            for (int it = 0; it < 4; ++it) {
                uint2 o;
                o.x = pk_bf16(prA[it].x, prA[it].y);
                o.y = pk_bf16(prA[it].z, prA[it].w);
                *(uint2*)&As[(a_row0 + it * 32) * BK + a_col] = o;
            }
        }
    };
    auto loadB = [&](int k0) {
        if constexpr (B_TRANSPOSE) {
#pragma unroll
            for (int it = 0; it < 8; ++it)
                prB[it] = *(const float4*)(Bf + (size_t)(k0 + bt_k) * N + (n0 + bt_n0 + it * 4));
        } else {
#pragma unroll
            for (int it = 0; it < 8; ++it)
                prB[it] = *(const float4*)(Bf + (size_t)(n0 + b_row0 + it * 32) * K + (k0 + b_col));
        }
    };
    auto storeB = [&](uint16_t* Bs) {
        if constexpr (B_TRANSPOSE) {
#pragma unroll
            for (int it = 0; it < 8; ++it) {
                const float* fp = (const float*)&prB[it];
#pragma unroll
                for (int r = 0; r < 4; ++r)
                    Bs[(bt_n0 + it * 4 + r) * BK + bt_k] = f32_to_bf16(fp[r]);
            }
        } else {
#pragma unroll
            for (int it = 0; it < 8; ++it) {
                uint2 o;
                o.x = pk_bf16(prB[it].x, prB[it].y);
                o.y = pk_bf16(prB[it].z, prB[it].w);
                *(uint2*)&Bs[(b_row0 + it * 32) * BK + b_col] = o;
            }
        }
    };
    auto compute = [&](const uint16_t* As, const uint16_t* Bs) {
        FragAB afr[4], bfr[4];
#pragma unroll
        for (int fm = 0; fm < 4; ++fm) {
            int row = wm * 64 + fm * 16 + lhalf;
            afr[fm].u[0] = *(const uint4*)&As[row * BK + lhi * 8];
            afr[fm].u[1] = *(const uint4*)&As[row * BK + lhi * 8 + 16];
        }
#pragma unroll
        for (int fn = 0; fn < 4; ++fn) {
            int colr = wn * 64 + fn * 16 + lhalf;
            bfr[fn].u[0] = *(const uint4*)&Bs[colr * BK + lhi * 8];
            bfr[fn].u[1] = *(const uint4*)&Bs[colr * BK + lhi * 8 + 16];
        }
#pragma unroll
        for (int fm = 0; fm < 4; ++fm)
#pragma unroll
            for (int fn = 0; fn < 4; ++fn)
                acc[fm][fn] = __builtin_amdgcn_wmma_f32_16x16x32_bf16(
                    false, afr[fm].v, false, bfr[fn].v,
                    (short)0, acc[fm][fn], false, false);
    };

    // ---- software-pipelined main loop (1 barrier / K-step) --------------
    const int nk = K / BK;
    loadB(0);
    if constexpr (A_TDM) {
#if USE_TDM
        if (wid == 0)
            tdm_load_tile_bf16(Ab + (size_t)m0 * K,
                               (uint32_t)(uintptr_t)&sA[0][0],
                               (uint32_t)(M - m0), (uint32_t)K);
#endif
    } else {
        loadA(0);
        storeA(sA[0]);
    }
    storeB(sB[0]);

    for (int i = 0; i < nk; ++i) {
        const int s = i & 1;
        const bool more = (i + 1 < nk);
        const int kn = (i + 1) * BK;
        if (more) {
            loadB(kn);
            if constexpr (!A_TDM) loadA(kn);
        }
#if USE_TDM
        if constexpr (A_TDM) {
            if (wid == 0) __builtin_amdgcn_s_wait_tensorcnt(0);  // buf s landed
        }
#endif
        __syncthreads();   // buf s visible; prior reads of buf s^1 retired
#if USE_TDM
        if constexpr (A_TDM) {
            if (wid == 0 && more)
                tdm_load_tile_bf16(Ab + (size_t)m0 * K + kn,
                                   (uint32_t)(uintptr_t)&sA[s ^ 1][0],
                                   (uint32_t)(M - m0), (uint32_t)K);
        }
#endif
        compute(sA[s], sB[s]);
        if (more) {
            storeB(sB[s ^ 1]);
            if constexpr (!A_TDM) storeA(sA[s ^ 1]);
        }
    }

    // ---- epilogue -------------------------------------------------------
#pragma unroll
    for (int fm = 0; fm < 4; ++fm) {
#pragma unroll
        for (int fn = 0; fn < 4; ++fn) {
#pragma unroll
            for (int r = 0; r < 8; ++r) {
                int mrow = m0 + wm * 64 + fm * 16 + r + (lhi << 3);
                int ncol = n0 + wn * 64 + fn * 16 + lhalf;
                float v = acc[fm][fn][r];
                if constexpr (MODE == MODE_S) {
                    v *= scale;
                    if (extra0[(size_t)bz * N + ncol] == 0.0f)
                        v = -4294967295.0f;                      // NEG_INF
                    ((uint16_t*)Obase)[zO + (size_t)mrow * N + ncol] = f32_to_bf16(v);
                } else if constexpr (MODE == MODE_PV) {
                    v += extra0[zO + (size_t)mrow * N + ncol];   // + queries
                    ((float*)Obase)[zO + (size_t)mrow * N + ncol] = v;
                } else if constexpr (MODE == MODE_F1) {
                    v += extra0[ncol];                           // + b1
                    v = fmaxf(v, 0.0f);                          // ReLU
                    ((uint16_t*)Obase)[(size_t)mrow * N + ncol] = f32_to_bf16(v);
                } else {                                          // MODE_F2
                    v += extra0[ncol] + extra1[(size_t)mrow * N + ncol];
                    ((float*)Obase)[(size_t)mrow * N + ncol] = v;
                }
            }
        }
    }
}

// ---------------------------------------------------------------------------
// Row softmax over Lk=2048: one WG per (b,i) row, 8 elems/thread.
// ---------------------------------------------------------------------------
__global__ __launch_bounds__(256)
void softmax_kernel(const uint16_t* __restrict__ S, uint16_t* __restrict__ P,
                    const float* __restrict__ qmask)
{
    __shared__ float red[256];
    const int tid = threadIdx.x;
    const size_t row = blockIdx.x;
    const uint16_t* s = S + row * (size_t)LK;

    uint4 r4 = *(const uint4*)(s + tid * 8);
    uint32_t u[4] = { r4.x, r4.y, r4.z, r4.w };
    float v[8];
#pragma unroll
    for (int j = 0; j < 4; ++j) {
        v[2 * j]     = __builtin_bit_cast(float, u[j] << 16);
        v[2 * j + 1] = __builtin_bit_cast(float, u[j] & 0xFFFF0000u);
    }
    float mx = v[0];
#pragma unroll
    for (int j = 1; j < 8; ++j) mx = fmaxf(mx, v[j]);
    red[tid] = mx; __syncthreads();
    for (int st = 128; st > 0; st >>= 1) {
        if (tid < st) red[tid] = fmaxf(red[tid], red[tid + st]);
        __syncthreads();
    }
    mx = red[0]; __syncthreads();

    float sum = 0.0f;
#pragma unroll
    for (int j = 0; j < 8; ++j) { v[j] = __expf(v[j] - mx); sum += v[j]; }
    red[tid] = sum; __syncthreads();
    for (int st = 128; st > 0; st >>= 1) {
        if (tid < st) red[tid] += red[tid + st];
        __syncthreads();
    }
    sum = red[0];

    const float sc = qmask[row] / sum;
    uint4 o;
    o.x = pk_bf16(v[0] * sc, v[1] * sc);
    o.y = pk_bf16(v[2] * sc, v[3] * sc);
    o.z = pk_bf16(v[4] * sc, v[5] * sc);
    o.w = pk_bf16(v[6] * sc, v[7] * sc);
    *(uint4*)(P + row * (size_t)LK + tid * 8) = o;
}

// ---------------------------------------------------------------------------
// LayerNorm over E=1024: one WG per row; 4 elems/thread. In-place safe.
// ---------------------------------------------------------------------------
__global__ __launch_bounds__(256)
void layernorm_kernel(const float* __restrict__ in, float* __restrict__ out,
                      const float* __restrict__ w, const float* __restrict__ bia)
{
    __shared__ float red[256];
    const int tid = threadIdx.x;
    const size_t row = blockIdx.x;
    const float* x = in + row * (size_t)EMB + tid * 4;
    float4 v = *(const float4*)x;

    red[tid] = v.x + v.y + v.z + v.w; __syncthreads();
    for (int st = 128; st > 0; st >>= 1) {
        if (tid < st) red[tid] += red[tid + st];
        __syncthreads();
    }
    const float mu = red[0] * (1.0f / EMB); __syncthreads();

    float dx = v.x - mu, dy = v.y - mu, dz = v.z - mu, dw = v.w - mu;
    red[tid] = dx * dx + dy * dy + dz * dz + dw * dw; __syncthreads();
    for (int st = 128; st > 0; st >>= 1) {
        if (tid < st) red[tid] += red[tid + st];
        __syncthreads();
    }
    const float rstd = rsqrtf(red[0] * (1.0f / EMB) + 1e-5f);

    const float4 wv = *(const float4*)(w + tid * 4);
    const float4 bv = *(const float4*)(bia + tid * 4);
    float4 o;
    o.x = dx * rstd * wv.x + bv.x;
    o.y = dy * rstd * wv.y + bv.y;
    o.z = dz * rstd * wv.z + bv.z;
    o.w = dw * rstd * wv.w + bv.w;
    *(float4*)(out + row * (size_t)EMB + tid * 4) = o;
}

// ---------------------------------------------------------------------------
extern "C" void kernel_launch(void* const* d_in, const int* in_sizes, int n_in,
                              void* d_out, int out_size, void* d_ws, size_t ws_size,
                              hipStream_t stream)
{
    const float* Q    = (const float*)d_in[0];
    const float* Kx   = (const float*)d_in[1];
    const float* qm   = (const float*)d_in[2];
    const float* km   = (const float*)d_in[3];
    const float* lnw  = (const float*)d_in[4];
    const float* lnb  = (const float*)d_in[5];
    const float* ln2w = (const float*)d_in[6];
    const float* ln2b = (const float*)d_in[7];
    const float* W1   = (const float*)d_in[8];
    const float* b1   = (const float*)d_in[9];
    const float* W2   = (const float*)d_in[10];
    const float* b2   = (const float*)d_in[11];
    float* out = (float*)d_out;

    char* ws = (char*)d_ws;
    uint16_t* S = (uint16_t*)(ws);                          // 64 MB
    uint16_t* P = (uint16_t*)(ws + ((size_t)64 << 20));     // 64 MB
    float*    X = (float*)   (ws + ((size_t)128 << 20));    // 64 MB
    float*    R2 = (float*)S;      // S dead after softmax
    uint16_t* H  = P;              // P dead after P@K

    const float scale = 1.0f / (sqrtf((float)EMB) + 1e-8f);
    const int Mrows = NB * LQ;     // 16384
    dim3 blk(256);

    // 1) S = scale * Q K^T  (+ key mask)
    gemm_wmma_kernel<MODE_S><<<dim3(LK / BN, LQ / BM, NB), blk, 0, stream>>>(
        Q, Kx, S, km, nullptr, LQ, LK, EMB, scale);
    // 2) P = softmax(S) * query_mask
    softmax_kernel<<<dim3(NB * LQ), blk, 0, stream>>>(S, P, qm);
    // 3) r1 = P K + Q          (A tiles streamed by TDM)
    gemm_wmma_kernel<MODE_PV><<<dim3(EMB / BN, LQ / BM, NB), blk, 0, stream>>>(
        P, Kx, X, Q, nullptr, LQ, EMB, LK, 1.0f);
    // 4) x = LN(r1)   (in place)
    layernorm_kernel<<<dim3(NB * LQ), blk, 0, stream>>>(X, X, lnw, lnb);
    // 5) H = relu(x W1^T + b1)
    gemm_wmma_kernel<MODE_F1><<<dim3(EMB / BN, Mrows / BM, 1), blk, 0, stream>>>(
        X, W1, H, b1, nullptr, Mrows, EMB, EMB, 1.0f);
    // 6) r2 = H W2^T + b2 + x  (A tiles streamed by TDM)
    gemm_wmma_kernel<MODE_F2><<<dim3(EMB / BN, Mrows / BM, 1), blk, 0, stream>>>(
        H, W2, R2, b2, X, Mrows, EMB, EMB, 1.0f);
    // 7) out = LN(r2)
    layernorm_kernel<<<dim3(NB * LQ), blk, 0, stream>>>(R2, out, ln2w, ln2b);
}
